// MultiHeadAttention_56667798503571
// MI455X (gfx1250) — matrix operand order repaired and checked by reading, not compiled, not run on hardware
//
#include <hip/hip_runtime.h>

// ---------------------------------------------------------------------------
// MI455X / gfx1250 multi-head attention, bf16 WMMA pipeline.
//   B=2, S=2048, D_MODEL=1024, HEAD=16, D_K=64
// Pass 0: one-shot fp32 -> bf16 conversion of X (q,k,v) and W (hardware cvt).
// Pass 1: projections via v_wmma_f32_16x16x32_bf16.
//         qh/kh stored [n][s][d] (K-contiguous = B-frag friendly for QK^T),
//         vh stored transposed [n][d][s] (B-frag friendly for PV).
//         1/sqrt(d_k) folded into the Q projection epilogue.
// Pass 2: flash attention. K/V blocks staged once per block into LDS via
//         async global->LDS DMA (ASYNCcnt), double-buffered; online softmax;
//         P transposed C-layout -> A-layout through a per-wave LDS tile.
// ---------------------------------------------------------------------------

typedef __attribute__((ext_vector_type(16))) __bf16 v16bf;
typedef __attribute__((ext_vector_type(8)))  __bf16 v8bf;
typedef __attribute__((ext_vector_type(8)))  float  v8f;
typedef int v4i_ __attribute__((vector_size(16)));   // <4 x i32> for async-DMA builtin

union BF16x16 { v16bf v; v8bf h[2]; __bf16 b[16]; };
union F32x8   { v8f   v; float f[8]; };

#if __has_builtin(__builtin_amdgcn_global_load_async_to_lds_b128)
#define ASYNC_LDS 1
#else
#define ASYNC_LDS 0
#endif

// stage 16 bytes global -> LDS (async DMA if available, else plain copy)
__device__ __forceinline__ void stage16(const __bf16* g, __bf16* l) {
#if ASYNC_LDS
  __builtin_amdgcn_global_load_async_to_lds_b128(
      (__attribute__((address_space(1))) v4i_*)g,
      (__attribute__((address_space(3))) v4i_*)l, 0, 0);
#else
  *reinterpret_cast<v8bf*>(l) = *reinterpret_cast<const v8bf*>(g);
#endif
}

template <int N>
__device__ __forceinline__ void wait_async() {
#if ASYNC_LDS
#if __has_builtin(__builtin_amdgcn_s_wait_asynccnt)
  __builtin_amdgcn_s_wait_asynccnt((short)N);
#else
  asm volatile("s_wait_asynccnt %0" ::"n"(N));
#endif
#endif
}

// ---------------------------------------------------------------------------
// Pass 0: streaming fp32 -> bf16 (8 elements / thread, b128 in, b128 out)
// ---------------------------------------------------------------------------
__global__ __launch_bounds__(256) void cvt_f32_to_bf16(
    const float* __restrict__ src, __bf16* __restrict__ dst, int n8)
{
  const int i = blockIdx.x * 256 + threadIdx.x;
  if (i >= n8) return;
  const float4* p = reinterpret_cast<const float4*>(src) + 2 * (size_t)i;
  const float4 a = p[0], b = p[1];
  v8bf o;
  o[0] = (__bf16)a.x; o[1] = (__bf16)a.y; o[2] = (__bf16)a.z; o[3] = (__bf16)a.w;
  o[4] = (__bf16)b.x; o[5] = (__bf16)b.y; o[6] = (__bf16)b.z; o[7] = (__bf16)b.w;
  reinterpret_cast<v8bf*>(dst)[i] = o;
}

// ---------------------------------------------------------------------------
// Pass 1: y = oscale * X(4096x1024) * W^T(1024x1024), bf16 in, head-split.
// Block = 4 waves, 64(M) x 64(N) tile; wave w owns M-tile w and all 4 N-tiles.
// v_trans==0: out[n][s][d] ; v_trans==1: out[n][d][s] (transposed V)
// ---------------------------------------------------------------------------
__global__ __launch_bounds__(128) void proj_bf16_wmma(
    const __bf16* __restrict__ X, const __bf16* __restrict__ W,
    __bf16* __restrict__ out, int v_trans, float oscale)
{
  const int lane = threadIdx.x & 31;
  const int wave = threadIdx.x >> 5;
  const int n16  = lane & 15;          // column / row-in-tile index
  const int g    = lane >> 4;          // half-wave group
  const int r0   = blockIdx.x * 64 + wave * 16;   // M base
  const int e0   = blockIdx.y * 64;               // N base

  F32x8 acc[4];
#pragma unroll
  for (int t = 0; t < 4; ++t)
#pragma unroll
    for (int i = 0; i < 8; ++i) acc[t].f[i] = 0.0f;

  const __bf16* xrow = X + (size_t)(r0 + n16) * 1024;

  for (int k0 = 0; k0 < 1024; k0 += 32) {
    // A fragment: row r0+n16, K = 16g..16g+15 contiguous
    BF16x16 a;
    a.h[0] = *reinterpret_cast<const v8bf*>(xrow + k0 + 16 * g);
    a.h[1] = *reinterpret_cast<const v8bf*>(xrow + k0 + 16 * g + 8);
#pragma unroll
    for (int t = 0; t < 4; ++t) {
      // B fragment: B[k][n] = W[e0+16t+n][k]; lane reads 16 contiguous k
      BF16x16 b;
      const __bf16* wrow = W + (size_t)(e0 + 16 * t + n16) * 1024 + k0 + 16 * g;
      b.h[0] = *reinterpret_cast<const v8bf*>(wrow);
      b.h[1] = *reinterpret_cast<const v8bf*>(wrow + 8);
      acc[t].v = __builtin_amdgcn_wmma_f32_16x16x32_bf16(
          false, a.v, false, b.v, (short)0, acc[t].v, false, false);
    }
  }

  // Epilogue: C/D layout -> lane holds column (e), rows M = vv + 8g
  const int rbase = r0 + 8 * g;
  const int bb    = rbase >> 11;        // batch
  const int sbase = rbase & 2047;       // sequence base (rows sbase+vv)
  if (!v_trans) {
#pragma unroll
    for (int t = 0; t < 4; ++t) {
      const int e = e0 + 16 * t + n16;
      const int n = bb * 16 + (e >> 6);
      const int d = e & 63;
      __bf16* o = out + ((size_t)n * 2048 + sbase) * 64 + d;
#pragma unroll
      for (int vv = 0; vv < 8; ++vv)
        o[(size_t)vv * 64] = (__bf16)(acc[t].f[vv] * oscale);
    }
  } else {
    // transposed store: 8 consecutive s values pack into one b128
#pragma unroll
    for (int t = 0; t < 4; ++t) {
      const int e = e0 + 16 * t + n16;
      const int n = bb * 16 + (e >> 6);
      const int d = e & 63;
      v8bf pk;
#pragma unroll
      for (int vv = 0; vv < 8; ++vv) pk[vv] = (__bf16)(acc[t].f[vv] * oscale);
      *reinterpret_cast<v8bf*>(out + ((size_t)n * 64 + d) * 2048 + sbase) = pk;
    }
  }
}

// ---------------------------------------------------------------------------
// Pass 2: flash attention over one head-batch slot n per blockIdx.y.
// Block = 4 waves x 16 query rows = 64 rows. Loop over 32 K/V columns / iter:
// K/V block staged once per BLOCK into LDS (async DMA, double buffered), all
// 4 waves build fragments from LDS -> 4x less L2 traffic than per-wave loads.
// Per iter per wave: 4 score WMMAs (K=64) + online softmax + 4 PV WMMAs.
// ---------------------------------------------------------------------------
__global__ __launch_bounds__(128) void flash_attn_wmma(
    const __bf16* __restrict__ QH, const __bf16* __restrict__ KH,
    const __bf16* __restrict__ VT, float* __restrict__ OUT)
{
  __shared__ __bf16 Klds[2][32][64];   // [buf][j-rel][d]     4 KB x2
  __shared__ __bf16 Vlds[2][64][32];   // [buf][d][j-rel]     4 KB x2
  __shared__ __bf16 Plds[4][16][32];   // per-wave P tile     4 KB

  const int tid  = threadIdx.x;
  const int lane = tid & 31;
  const int wave = tid >> 5;
  const int n16  = lane & 15;
  const int g    = lane >> 4;
  const int n    = blockIdx.y;                 // 0..31 (= b*16 + h)
  const int s0   = blockIdx.x * 64 + wave * 16;

  const __bf16* qrow  = QH + ((size_t)n * 2048 + s0 + n16) * 64;
  const __bf16* kbase = KH + (size_t)n * 2048 * 64;
  const __bf16* vbase = VT + (size_t)n * 64 * 2048;

  // stage one 32-column K/V block into LDS buffer `buf` (4 x b128 per thread)
  auto stageKV = [&](int buf, int j0) {
    const int c0 = tid, c1 = tid + 128;
    stage16(kbase + (size_t)(j0 + (c0 >> 3)) * 64 + (c0 & 7) * 8,
            &Klds[buf][c0 >> 3][(c0 & 7) * 8]);
    stage16(kbase + (size_t)(j0 + (c1 >> 3)) * 64 + (c1 & 7) * 8,
            &Klds[buf][c1 >> 3][(c1 & 7) * 8]);
    stage16(vbase + (size_t)(c0 >> 2) * 2048 + j0 + (c0 & 3) * 8,
            &Vlds[buf][c0 >> 2][(c0 & 3) * 8]);
    stage16(vbase + (size_t)(c1 >> 2) * 2048 + j0 + (c1 & 3) * 8,
            &Vlds[buf][c1 >> 2][(c1 & 3) * 8]);
  };

  // Q A-fragments for K(d) = 0..31 and 32..63, resident for whole loop
  // (Q was pre-scaled by 1/sqrt(d_k) in the projection.)
  BF16x16 qa[2];
#pragma unroll
  for (int kk = 0; kk < 2; ++kk) {
    qa[kk].h[0] = *reinterpret_cast<const v8bf*>(qrow + 32 * kk + 16 * g);
    qa[kk].h[1] = *reinterpret_cast<const v8bf*>(qrow + 32 * kk + 16 * g + 8);
  }

  F32x8 acc[4];
#pragma unroll
  for (int t = 0; t < 4; ++t)
#pragma unroll
    for (int i = 0; i < 8; ++i) acc[t].f[i] = 0.0f;
  float mrun[8], lrun[8];
#pragma unroll
  for (int i = 0; i < 8; ++i) { mrun[i] = -3.0e38f; lrun[i] = 0.0f; }

  stageKV(0, 0);                       // prologue DMA

  for (int j0 = 0; j0 < 2048; j0 += 32) {
    const int cur = (j0 >> 5) & 1;
    if (j0 + 32 < 2048) {              // uniform branch
      stageKV(cur ^ 1, j0 + 32);       // prefetch next block (async)
      wait_async<4>();                 // current block's 4 DMAs done
    } else {
      wait_async<0>();
    }
    __syncthreads();                   // LDS visible across waves

    // ---- scores: two 16x16 tiles (cols j0..j0+15, j0+16..j0+31) ----
    F32x8 sc[2];
#pragma unroll
    for (int jt = 0; jt < 2; ++jt) {
      F32x8 c;
#pragma unroll
      for (int i = 0; i < 8; ++i) c.f[i] = 0.0f;
#pragma unroll
      for (int kk = 0; kk < 2; ++kk) {
        BF16x16 kb;   // B[k][col] = Klds[cur][16jt+col][32kk + k]
        const __bf16* kr = &Klds[cur][16 * jt + n16][32 * kk + 16 * g];
        kb.h[0] = *reinterpret_cast<const v8bf*>(kr);
        kb.h[1] = *reinterpret_cast<const v8bf*>(kr + 8);
        c.v = __builtin_amdgcn_wmma_f32_16x16x32_bf16(
            false, qa[kk].v, false, kb.v, (short)0, c.v, false, false);
      }
      sc[jt] = c;
    }

    // ---- online softmax (scale already folded into Q) ----
    float corr[8];
#pragma unroll
    for (int vv = 0; vv < 8; ++vv) {
      float a0 = sc[0].f[vv];
      float a1 = sc[1].f[vv];
      float m  = fmaxf(a0, a1);
      m = fmaxf(m, __shfl_xor(m, 1, 16));
      m = fmaxf(m, __shfl_xor(m, 2, 16));
      m = fmaxf(m, __shfl_xor(m, 4, 16));
      m = fmaxf(m, __shfl_xor(m, 8, 16));
      const float mn = fmaxf(mrun[vv], m);
      corr[vv] = __expf(mrun[vv] - mn);
      mrun[vv] = mn;
      const float p0 = __expf(a0 - mn);
      const float p1 = __expf(a1 - mn);
      // C/D layout: lane holds column n16(+16), row vv + 8g
      Plds[wave][vv + 8 * g][n16]      = (__bf16)p0;
      Plds[wave][vv + 8 * g][16 + n16] = (__bf16)p1;
      float s = p0 + p1;
      s += __shfl_xor(s, 1, 16);
      s += __shfl_xor(s, 2, 16);
      s += __shfl_xor(s, 4, 16);
      s += __shfl_xor(s, 8, 16);
      lrun[vv] = lrun[vv] * corr[vv] + s;
#pragma unroll
      for (int t = 0; t < 4; ++t) acc[t].f[vv] *= corr[vv];
    }
    asm volatile("" ::: "memory");   // keep LDS stores before transposed reads

    // ---- re-read P as A-fragment (lane = row) : wave-internal transpose ----
    BF16x16 pa;
    const __bf16* pr = &Plds[wave][n16][0];
    pa.h[0] = *reinterpret_cast<const v8bf*>(pr + 16 * g);       // K = 16g..
    pa.h[1] = *reinterpret_cast<const v8bf*>(pr + 16 * g + 8);   // K = 16g+8..

    // ---- PV: 4 N-tiles of 16 output features ----
#pragma unroll
    for (int t = 0; t < 4; ++t) {
      BF16x16 vb;   // B[k][d] = Vlds[cur][16t + d][k] : contiguous in k
      const __bf16* vr = &Vlds[cur][16 * t + n16][16 * g];
      vb.h[0] = *reinterpret_cast<const v8bf*>(vr);
      vb.h[1] = *reinterpret_cast<const v8bf*>(vr + 8);
      acc[t].v = __builtin_amdgcn_wmma_f32_16x16x32_bf16(
          false, pa.v, false, vb.v, (short)0, acc[t].v, false, false);
    }
    __syncthreads();   // all waves done with buffer `cur` before it is re-DMAed
  }

  // ---- epilogue: out[b=n%2][s][ (n/2)*64 + d ] = acc / rowsum ----
  const int b2   = n & 1;
  const int cofs = (n >> 1) * 64;
#pragma unroll
  for (int vv = 0; vv < 8; ++vv) {
    const float inv = 1.0f / lrun[vv];
    const int s = s0 + vv + 8 * g;
    float* orow = OUT + ((size_t)b2 * 2048 + s) * 1024 + cofs;
#pragma unroll
    for (int t = 0; t < 4; ++t)
      orow[16 * t + n16] = acc[t].f[vv] * inv;
  }
}

// ---------------------------------------------------------------------------
extern "C" void kernel_launch(void* const* d_in, const int* in_sizes, int n_in,
                              void* d_out, int out_size, void* d_ws, size_t ws_size,
                              hipStream_t stream) {
  const float* q  = (const float*)d_in[0];
  const float* k  = (const float*)d_in[1];
  const float* v  = (const float*)d_in[2];
  const float* Wq = (const float*)d_in[3];
  const float* Wk = (const float*)d_in[4];
  const float* Wv = (const float*)d_in[5];
  float* out = (float*)d_out;

  // workspace layout (bf16 elements):
  //   qh, kh (n,s,d), vhT (n,d,s): 3 * 4M elems  (24 MB)
  //   xq, xk, xv bf16 copies:      3 * 4M elems  (24 MB)
  //   Wq, Wk, Wv bf16 copies:      3 * 1M elems  ( 6 MB)
  const size_t SL = (size_t)32 * 2048 * 64;   // 4M
  const size_t WL = (size_t)1024 * 1024;      // 1M
  __bf16* qh  = (__bf16*)d_ws;
  __bf16* kh  = qh + SL;
  __bf16* vhT = kh + SL;
  __bf16* xq  = vhT + SL;
  __bf16* xk  = xq + SL;
  __bf16* xv  = xk + SL;
  __bf16* wq  = xv + SL;
  __bf16* wk  = wq + WL;
  __bf16* wv  = wk + WL;

  // Pass 0: fp32 -> bf16 (hardware cvt), streaming
  const int nX8 = (int)(SL / 8), nW8 = (int)(WL / 8);
  cvt_f32_to_bf16<<<dim3((nX8 + 255) / 256), dim3(256), 0, stream>>>(q,  xq, nX8);
  cvt_f32_to_bf16<<<dim3((nX8 + 255) / 256), dim3(256), 0, stream>>>(k,  xk, nX8);
  cvt_f32_to_bf16<<<dim3((nX8 + 255) / 256), dim3(256), 0, stream>>>(v,  xv, nX8);
  cvt_f32_to_bf16<<<dim3((nW8 + 255) / 256), dim3(256), 0, stream>>>(Wq, wq, nW8);
  cvt_f32_to_bf16<<<dim3((nW8 + 255) / 256), dim3(256), 0, stream>>>(Wk, wk, nW8);
  cvt_f32_to_bf16<<<dim3((nW8 + 255) / 256), dim3(256), 0, stream>>>(Wv, wv, nW8);

  // Pass 1: projections (Q pre-scaled by 1/sqrt(d_k) = 0.125)
  dim3 gp(64, 16), bp(128);
  proj_bf16_wmma<<<gp, bp, 0, stream>>>(xq, wq, qh, 0, 0.125f);
  proj_bf16_wmma<<<gp, bp, 0, stream>>>(xk, wk, kh, 0, 1.0f);
  proj_bf16_wmma<<<gp, bp, 0, stream>>>(xv, wv, vhT, 1, 1.0f);

  // Pass 2: fused attention
  dim3 ga(32, 32), ba(128);
  flash_attn_wmma<<<ga, ba, 0, stream>>>(qh, kh, vhT, out);
}
